// MyGML2_62929860821274
// MI455X (gfx1250) — matrix operand order repaired and checked by reading, not compiled, not run on hardware
//
#include <hip/hip_runtime.h>
#include <hip/hip_bf16.h>
#include <math.h>

// ---------------------------------------------------------------------------
// Model constants
// ---------------------------------------------------------------------------
#define SEQ   512
#define DMODEL 4096
#define NHEAD 32
#define NKVH  2
#define HDIM  128
#define ROTD  64
#define GSZ   128
#define FFH   13696
#define FF2H  27392
#define VOCAB 65024
#define QKVO  4608

typedef _Float16 v16h __attribute__((ext_vector_type(16)));
typedef float    v8f  __attribute__((ext_vector_type(8)));
typedef unsigned int u32x4 __attribute__((ext_vector_type(4)));
typedef float    f32x4 __attribute__((ext_vector_type(4)));

union FragH { v16h h; u32x4 q[2]; };
union Pack8 { _Float16 h[8]; u32x4 q; };

// pack two f32 -> two f16 in one dword (enables ds_store_b32)
__device__ __forceinline__ unsigned pack2h(float lo, float hi) {
    union { _Float16 h[2]; unsigned u; } p;
    p.h[0] = (_Float16)lo;
    p.h[1] = (_Float16)hi;
    return p.u;
}

// A-fragment (16x32 f16) per ISA 7.12.2:
//  lanes 0-15 : halves 0-7 -> K 0..7,  halves 8-15 -> K 16..23
//  lanes 16-31: halves 0-7 -> K 8..15, halves 8-15 -> K 24..31
__device__ __forceinline__ v16h lds_frag_a(const _Float16* rowPtr, int kc, int lane) {
    const int ko = (lane & 16) ? 8 : 0;
    FragH f;
    f.q[0] = *(const u32x4*)(rowPtr + kc * 32 + ko);
    f.q[1] = *(const u32x4*)(rowPtr + kc * 32 + ko + 16);
    return f.h;
}

// B-fragment (32x16 f16), B stored K-contiguous per output column n:
//  lanes 0-15 : halves 0-7 -> K 0..7,  halves 8-15 -> K 16..23
//  lanes 16-31: halves 0-7 -> K 8..15, halves 8-15 -> K 24..31
__device__ __forceinline__ v16h lds_frag_b(const _Float16* rowPtr, int kc, int lane) {
    const int ks = (lane & 16) ? 8 : 0;
    FragH f;
    f.q[0] = *(const u32x4*)(rowPtr + kc * 32 + ks);
    f.q[1] = *(const u32x4*)(rowPtr + kc * 32 + ks + 16);
    return f.h;
}

__device__ __forceinline__ v8f wmma_f16(v16h a, v16h b, v8f c) {
    // D = A*B + C, f32 accumulate.  emits v_wmma_f32_16x16x32_f16
    return __builtin_amdgcn_wmma_f32_16x16x32_f16(false, a, false, b, (short)0, c,
                                                  false, false);
}

#define LDP 136   // padded LDS row (halves): 272B, 16B aligned, bank-conflict free

// ---------------------------------------------------------------------------
// Quantized-weight WMMA GEMM: Y[M,N] = X[M,K](f16) @ deq(Wq)[K,N] (+bias)(+res)
//   Wq int8 [K,N], scales f32 [K/128,N], zeros int8 [K/128,N]
//   Block: 256 thr (8 waves), tile 64(M) x 64(N), K-step 128 (one quant group)
// ---------------------------------------------------------------------------
template<bool HAS_BIAS, bool HAS_RES>
__launch_bounds__(256)
__global__ void gemm_q8_wmma(const _Float16* __restrict__ X,
                             const signed char* __restrict__ Wq,
                             const float* __restrict__ Wsc,
                             const signed char* __restrict__ Wz,
                             const float* __restrict__ bias,
                             const float* __restrict__ res,
                             float* __restrict__ Y,
                             int M, int N, int K) {
    __shared__ __align__(16) _Float16 As[64 * LDP];
    __shared__ __align__(16) _Float16 Bt[64 * LDP];   // transposed: [n][k]
    __shared__ float sS[64];
    __shared__ float sZ[64];

    const int tid  = threadIdx.x;
    const int lane = tid & 31;
    const int wave = tid >> 5;
    const int mt   = wave >> 1;          // 0..3   (16-row tile)
    const int ntB  = (wave & 1) * 2;     // 0 or 2 (two 16-col tiles per wave)
    const int n0   = blockIdx.x * 64;
    const int m0   = blockIdx.y * 64;

    v8f acc0 = {0.f,0.f,0.f,0.f,0.f,0.f,0.f,0.f};
    v8f acc1 = {0.f,0.f,0.f,0.f,0.f,0.f,0.f,0.f};

    const int nGroups = K >> 7;
    for (int g = 0; g < nGroups; ++g) {
        const int k0 = g << 7;
        __syncthreads();
        if (tid < 64) {
            sS[tid] = Wsc[(size_t)g * N + n0 + tid];
            sZ[tid] = (float)Wz[(size_t)g * N + n0 + tid];
        }
        // X tile: 64 rows x 128 halves (coalesced b128)
        {
            const int tr = tid >> 4;
            const int tc = (tid & 15) << 3;
#pragma unroll
            for (int p = 0; p < 4; ++p) {
                const int m = tr + (p << 4);
                *(u32x4*)&As[m * LDP + tc] =
                    *(const u32x4*)(X + (size_t)(m0 + m) * K + k0 + tc);
            }
        }
        // prefetch next weight K-group while we dequant this one
        if (g + 1 < nGroups)
            __builtin_prefetch(Wq + (size_t)(k0 + 128 + (tid >> 4)) * N + n0 +
                               ((tid & 15) << 2), 0, 1);
        __syncthreads();
        // dequant W tile [128k x 64n] int8 -> f16 transposed [n][k],
        // two adjacent k packed per dword LDS store
        {
            const int kk = (tid >> 4) << 1;     // even k base: 0,2,..,30
            const int nq = (tid & 15) << 2;     // 4 columns per thread
#pragma unroll
            for (int p = 0; p < 4; ++p) {
                const int k = kk + (p << 5);    // 32 k-rows per pass
                const int wa = *(const int*)(Wq + (size_t)(k0 + k) * N + n0 + nq);
                const int wb = *(const int*)(Wq + (size_t)(k0 + k + 1) * N + n0 + nq);
#pragma unroll
                for (int j = 0; j < 4; ++j) {
                    const float z = sZ[nq + j];
                    const float s = sS[nq + j];
                    const float qa = (float)((wa >> (8 * j)) & 0xFF);   // values 0..15
                    const float qb = (float)((wb >> (8 * j)) & 0xFF);
                    *(unsigned*)&Bt[(nq + j) * LDP + k] =
                        pack2h((qa - z) * s, (qb - z) * s);
                }
            }
        }
        __syncthreads();
#pragma unroll
        for (int kc = 0; kc < 4; ++kc) {
            const v16h a  = lds_frag_a(&As[(mt * 16 + (lane & 15)) * LDP], kc, lane);
            const v16h b0 = lds_frag_b(&Bt[((ntB    ) * 16 + (lane & 15)) * LDP], kc, lane);
            const v16h b1 = lds_frag_b(&Bt[((ntB + 1) * 16 + (lane & 15)) * LDP], kc, lane);
            acc0 = wmma_f16(a, b0, acc0);
            acc1 = wmma_f16(a, b1, acc1);
        }
    }
    // C/D layout: VGPR r -> M = r + 8*(lane>=16), N = lane&15
    const int nl = lane & 15;
    const int mh = (lane >> 4) << 3;
#pragma unroll
    for (int r = 0; r < 8; ++r) {
        const int gm = m0 + mt * 16 + mh + r;
        const size_t ro = (size_t)gm * N;
        {
            const int gn = n0 + ntB * 16 + nl;
            float v = acc0[r];
            if (HAS_BIAS) v += bias[gn];
            if (HAS_RES)  v += res[ro + gn];
            Y[ro + gn] = v;
        }
        {
            const int gn = n0 + (ntB + 1) * 16 + nl;
            float v = acc1[r];
            if (HAS_BIAS) v += bias[gn];
            if (HAS_RES)  v += res[ro + gn];
            Y[ro + gn] = v;
        }
    }
}

// ---------------------------------------------------------------------------
// fp32-weight WMMA GEMM (logits): Y = X(f16) @ W(f32->f16 in LDS)
// ---------------------------------------------------------------------------
__launch_bounds__(256)
__global__ void gemm_f32w_wmma(const _Float16* __restrict__ X,
                               const float* __restrict__ W,
                               float* __restrict__ Y,
                               int M, int N, int K) {
    __shared__ __align__(16) _Float16 As[64 * LDP];
    __shared__ __align__(16) _Float16 Bt[64 * LDP];

    const int tid  = threadIdx.x;
    const int lane = tid & 31;
    const int wave = tid >> 5;
    const int mt   = wave >> 1;
    const int ntB  = (wave & 1) * 2;
    const int n0   = blockIdx.x * 64;
    const int m0   = blockIdx.y * 64;

    v8f acc0 = {0.f,0.f,0.f,0.f,0.f,0.f,0.f,0.f};
    v8f acc1 = {0.f,0.f,0.f,0.f,0.f,0.f,0.f,0.f};

    for (int k0 = 0; k0 < K; k0 += 128) {
        __syncthreads();
        {
            const int tr = tid >> 4;
            const int tc = (tid & 15) << 3;
#pragma unroll
            for (int p = 0; p < 4; ++p) {
                const int m = tr + (p << 4);
                *(u32x4*)&As[m * LDP + tc] =
                    *(const u32x4*)(X + (size_t)(m0 + m) * K + k0 + tc);
            }
        }
        if (k0 + 128 < K)
            __builtin_prefetch(W + (size_t)(k0 + 128 + (tid >> 4)) * N + n0 +
                               ((tid & 15) << 2), 0, 1);
        // W tile f32 -> f16 transposed, two adjacent k packed per dword store
        {
            const int kk = (tid >> 4) << 1;
            const int nq = (tid & 15) << 2;
#pragma unroll
            for (int p = 0; p < 4; ++p) {
                const int k = kk + (p << 5);
                const f32x4 wa = *(const f32x4*)(W + (size_t)(k0 + k) * N + n0 + nq);
                const f32x4 wb = *(const f32x4*)(W + (size_t)(k0 + k + 1) * N + n0 + nq);
#pragma unroll
                for (int j = 0; j < 4; ++j)
                    *(unsigned*)&Bt[(nq + j) * LDP + k] = pack2h(wa[j], wb[j]);
            }
        }
        __syncthreads();
#pragma unroll
        for (int kc = 0; kc < 4; ++kc) {
            const v16h a  = lds_frag_a(&As[(mt * 16 + (lane & 15)) * LDP], kc, lane);
            const v16h b0 = lds_frag_b(&Bt[((ntB    ) * 16 + (lane & 15)) * LDP], kc, lane);
            const v16h b1 = lds_frag_b(&Bt[((ntB + 1) * 16 + (lane & 15)) * LDP], kc, lane);
            acc0 = wmma_f16(a, b0, acc0);
            acc1 = wmma_f16(a, b1, acc1);
        }
    }
    const int nl = lane & 15;
    const int mh = (lane >> 4) << 3;
#pragma unroll
    for (int r = 0; r < 8; ++r) {
        const int gm = m0 + mt * 16 + mh + r;
        const size_t ro = (size_t)gm * N;
        Y[ro + n0 + ntB * 16 + nl]       = acc0[r];
        Y[ro + (ntB + 1) * 16 + n0 + nl] = acc1[r];
    }
}

// ---------------------------------------------------------------------------
// Attention: scores[h,q,k] = (Q.Kt)/sqrt(HD), causal masked
// grid (ntile=8, mtile=8, head=32)
// ---------------------------------------------------------------------------
__launch_bounds__(256)
__global__ void attn_scores_wmma(const _Float16* __restrict__ Qf,
                                 const _Float16* __restrict__ Kf,
                                 float* __restrict__ scores) {
    __shared__ __align__(16) _Float16 Qs[64 * LDP];
    __shared__ __align__(16) _Float16 Ks[64 * LDP];   // [key][hd] == Bt for QK^T

    const int tid  = threadIdx.x;
    const int lane = tid & 31;
    const int wave = tid >> 5;
    const int mt   = wave >> 1;
    const int ntB  = (wave & 1) * 2;
    const int head = blockIdx.z;
    const int kvh  = head >> 4;                 // GQA: 16 q-heads per kv-head
    const int m0   = blockIdx.y * 64;
    const int n0   = blockIdx.x * 64;

    {
        const int tr = tid >> 4;
        const int tc = (tid & 15) << 3;
#pragma unroll
        for (int p = 0; p < 4; ++p) {
            const int m = tr + (p << 4);
            *(u32x4*)&Qs[m * LDP + tc] =
                *(const u32x4*)(Qf + ((size_t)head * SEQ + m0 + m) * HDIM + tc);
            *(u32x4*)&Ks[m * LDP + tc] =
                *(const u32x4*)(Kf + ((size_t)kvh * SEQ + n0 + m) * HDIM + tc);
        }
    }
    __syncthreads();

    v8f acc0 = {0.f,0.f,0.f,0.f,0.f,0.f,0.f,0.f};
    v8f acc1 = {0.f,0.f,0.f,0.f,0.f,0.f,0.f,0.f};
#pragma unroll
    for (int kc = 0; kc < 4; ++kc) {            // HD = 128 = 4*32
        const v16h a  = lds_frag_a(&Qs[(mt * 16 + (lane & 15)) * LDP], kc, lane);
        const v16h b0 = lds_frag_b(&Ks[((ntB    ) * 16 + (lane & 15)) * LDP], kc, lane);
        const v16h b1 = lds_frag_b(&Ks[((ntB + 1) * 16 + (lane & 15)) * LDP], kc, lane);
        acc0 = wmma_f16(a, b0, acc0);
        acc1 = wmma_f16(a, b1, acc1);
    }
    const float sc = 0.08838834764831845f;      // 1/sqrt(128)
    const int nl = lane & 15;
    const int mh = (lane >> 4) << 3;
#pragma unroll
    for (int r = 0; r < 8; ++r) {
        const int gm = m0 + mt * 16 + mh + r;
        float* row = scores + ((size_t)head * SEQ + gm) * SEQ;
        {
            const int gn = n0 + ntB * 16 + nl;
            row[gn] = (gn > gm) ? -1.0e9f : acc0[r] * sc;
        }
        {
            const int gn = n0 + (ntB + 1) * 16 + nl;
            row[gn] = (gn > gm) ? -1.0e9f : acc1[r] * sc;
        }
    }
}

// softmax over 512 elements, in place; one block per row
__launch_bounds__(256)
__global__ void softmax512(float* __restrict__ scores) {
    float* p = scores + (size_t)blockIdx.x * SEQ;
    const int t = threadIdx.x;
    const float a = p[t], b = p[t + 256];
    __shared__ float red[256];
    red[t] = fmaxf(a, b);
    __syncthreads();
    for (int o = 128; o > 0; o >>= 1) {
        if (t < o) red[t] = fmaxf(red[t], red[t + o]);
        __syncthreads();
    }
    const float m = red[0];
    __syncthreads();
    const float ea = expf(a - m), eb = expf(b - m);
    red[t] = ea + eb;
    __syncthreads();
    for (int o = 128; o > 0; o >>= 1) {
        if (t < o) red[t] += red[t + o];
        __syncthreads();
    }
    const float inv = 1.0f / red[0];
    p[t] = ea * inv;
    p[t + 256] = eb * inv;
}

// ctx[q, head*128+hd] = sum_key P[h,q,key] * V[kvh,key,hd]
// grid (1, mtile=8, head=32); BN=128 (full head dim), BK=128 keys
__launch_bounds__(256)
__global__ void attn_pv_wmma(const float* __restrict__ P,
                             const _Float16* __restrict__ Vf,
                             _Float16* __restrict__ ctx) {
    __shared__ __align__(16) _Float16 Ps[64 * LDP];
    __shared__ __align__(16) _Float16 Vt[128 * LDP];   // [hd][key]

    const int tid  = threadIdx.x;
    const int lane = tid & 31;
    const int wave = tid >> 5;
    const int mt   = wave >> 1;           // 0..3
    const int ntB  = (wave & 1) * 4;      // 0 or 4 (four 16-col tiles -> 128)
    const int head = blockIdx.z;
    const int kvh  = head >> 4;
    const int m0   = blockIdx.y * 64;

    v8f acc[4];
#pragma unroll
    for (int i = 0; i < 4; ++i) acc[i] = (v8f){0.f,0.f,0.f,0.f,0.f,0.f,0.f,0.f};

    for (int k0 = 0; k0 < SEQ; k0 += 128) {
        __syncthreads();
        // P tile 64q x 128key (f32 -> f16, packed b128 LDS store)
        {
            const int tr = tid >> 4;
            const int tc = (tid & 15) << 3;
#pragma unroll
            for (int p2 = 0; p2 < 4; ++p2) {
                const int m = tr + (p2 << 4);
                const float* src = P + ((size_t)head * SEQ + m0 + m) * SEQ + k0 + tc;
                const f32x4 w0 = *(const f32x4*)(src);
                const f32x4 w1 = *(const f32x4*)(src + 4);
                Pack8 pk;
#pragma unroll
                for (int j = 0; j < 4; ++j) {
                    pk.h[j]     = (_Float16)w0[j];
                    pk.h[4 + j] = (_Float16)w1[j];
                }
                *(u32x4*)&Ps[m * LDP + tc] = pk.q;
            }
        }
        // V tile transposed: [hd][key]
        {
            const int ky = tid >> 4;
            const int h8 = (tid & 15) << 3;
#pragma unroll
            for (int p2 = 0; p2 < 8; ++p2) {
                const int kk = ky + (p2 << 4);
                u32x4 v4 = *(const u32x4*)(Vf + ((size_t)kvh * SEQ + k0 + kk) * HDIM + h8);
                const _Float16* hv = (const _Float16*)&v4;
#pragma unroll
                for (int j = 0; j < 8; ++j) Vt[(h8 + j) * LDP + kk] = hv[j];
            }
        }
        __syncthreads();
#pragma unroll
        for (int kc = 0; kc < 4; ++kc) {
            const v16h a = lds_frag_a(&Ps[(mt * 16 + (lane & 15)) * LDP], kc, lane);
#pragma unroll
            for (int t2 = 0; t2 < 4; ++t2) {
                const v16h b = lds_frag_b(&Vt[((ntB + t2) * 16 + (lane & 15)) * LDP], kc, lane);
                acc[t2] = wmma_f16(a, b, acc[t2]);
            }
        }
    }
    const int nl = lane & 15;
    const int mh = (lane >> 4) << 3;
#pragma unroll
    for (int r = 0; r < 8; ++r) {
        const int gq = m0 + mt * 16 + mh + r;
#pragma unroll
        for (int t2 = 0; t2 < 4; ++t2) {
            const int hd = (ntB + t2) * 16 + nl;
            ctx[(size_t)gq * DMODEL + head * HDIM + hd] = (_Float16)acc[t2][r];
        }
    }
}

// ---------------------------------------------------------------------------
// Small elementwise kernels
// ---------------------------------------------------------------------------
__launch_bounds__(256)
__global__ void embed_gather(const int* __restrict__ ids,
                             const float* __restrict__ emb,
                             float* __restrict__ h) {
    const int s = blockIdx.x;
    const float* src = emb + (size_t)ids[s] * DMODEL;
    float* dst = h + (size_t)s * DMODEL;
    for (int i = threadIdx.x * 4; i < DMODEL; i += 256 * 4)
        *(f32x4*)(dst + i) = *(const f32x4*)(src + i);
}

__launch_bounds__(256)
__global__ void rmsnorm_f16(const float* __restrict__ x,
                            const float* __restrict__ w,
                            _Float16* __restrict__ y) {
    const int row = blockIdx.x;
    const float* xr = x + (size_t)row * DMODEL;
    float ss = 0.f;
    for (int i = threadIdx.x; i < DMODEL; i += 256) { const float v = xr[i]; ss += v * v; }
    __shared__ float red[256];
    red[threadIdx.x] = ss;
    __syncthreads();
    for (int o = 128; o > 0; o >>= 1) {
        if (threadIdx.x < o) red[threadIdx.x] += red[threadIdx.x + o];
        __syncthreads();
    }
    const float inv = rsqrtf(red[0] * (1.0f / DMODEL) + 1e-5f);
    _Float16* yr = y + (size_t)row * DMODEL;
    for (int i = threadIdx.x; i < DMODEL; i += 256)
        yr[i] = (_Float16)(w[i] * xr[i] * inv);
}

// split fp32 qkv -> f16 Q[32,S,128], K[2,S,128] (rope on Q,K), V[2,S,128]
__launch_bounds__(64)
__global__ void rope_split(const float* __restrict__ qkv,
                           _Float16* __restrict__ Qf,
                           _Float16* __restrict__ Kf,
                           _Float16* __restrict__ Vf) {
    const int s = blockIdx.y;
    const int u = blockIdx.x;            // 0..31 q-head, 32..33 k, 34..35 v
    const int e = threadIdx.x;           // 0..63, two elements each
    const float* src;
    _Float16* dst;
    bool doRope;
    if (u < 32)      { src = qkv + (size_t)s * QKVO + u * HDIM;
                       dst = Qf + ((size_t)u * SEQ + s) * HDIM;        doRope = true; }
    else if (u < 34) { const int j = u - 32;
                       src = qkv + (size_t)s * QKVO + 4096 + j * HDIM;
                       dst = Kf + ((size_t)j * SEQ + s) * HDIM;        doRope = true; }
    else             { const int j = u - 34;
                       src = qkv + (size_t)s * QKVO + 4352 + j * HDIM;
                       dst = Vf + ((size_t)j * SEQ + s) * HDIM;        doRope = false; }
    int d0, d1;
    if (e < 32) { d0 = 2 * e; d1 = d0 + 1; }
    else        { d0 = 64 + 2 * (e - 32); d1 = d0 + 1; }
    float x0 = src[d0], x1 = src[d1];
    if (doRope && e < 32) {
        const float th = (float)s * powf(10000.f, -(float)(2 * e) / (float)ROTD);
        const float c = cosf(th), sn = sinf(th);
        const float r0 = x0 * c - x1 * sn;
        const float r1 = x1 * c + x0 * sn;
        x0 = r0; x1 = r1;
    }
    dst[d0] = (_Float16)x0;
    dst[d1] = (_Float16)x1;
}

__launch_bounds__(256)
__global__ void silu_gate(const float* __restrict__ ff, _Float16* __restrict__ g) {
    const size_t i = (size_t)blockIdx.x * 256 + threadIdx.x;
    if (i >= (size_t)SEQ * FFH) return;
    const size_t m = i / FFH, j = i % FFH;
    const float a = ff[m * FF2H + j];
    const float b = ff[m * FF2H + FFH + j];
    g[i] = (_Float16)((a / (1.0f + expf(-a))) * b);
}

// ---------------------------------------------------------------------------
// Orchestration
// ---------------------------------------------------------------------------
extern "C" void kernel_launch(void* const* d_in, const int* in_sizes, int n_in,
                              void* d_out, int out_size, void* d_ws, size_t ws_size,
                              hipStream_t stream) {
    (void)in_sizes; (void)n_in; (void)out_size; (void)ws_size;
    const int*   ids   = (const int*)d_in[0];
    /* position_ids (d_in[1]) == arange; folded into rope */
    const float* emb   = (const float*)d_in[2];
    const float* ln1   = (const float*)d_in[3];
    const float* ln2   = (const float*)d_in[4];
    const float* lnf   = (const float*)d_in[5];
    const signed char* qkv_q = (const signed char*)d_in[6];
    const float*       qkv_s = (const float*)d_in[7];
    const signed char* qkv_z = (const signed char*)d_in[8];
    const float*       qkv_b = (const float*)d_in[9];
    const signed char* dn_q  = (const signed char*)d_in[10];
    const float*       dn_s  = (const float*)d_in[11];
    const signed char* dn_z  = (const signed char*)d_in[12];
    const signed char* h4_q  = (const signed char*)d_in[13];
    const float*       h4_s  = (const float*)d_in[14];
    const signed char* h4_z  = (const signed char*)d_in[15];
    const signed char* fh_q  = (const signed char*)d_in[16];
    const float*       fh_s  = (const float*)d_in[17];
    const signed char* fh_z  = (const signed char*)d_in[18];
    const float*       outw  = (const float*)d_in[19];

    // workspace carve-up (256B aligned)
    char* ws = (char*)d_ws;
    size_t off = 0;
    auto alloc = [&](size_t bytes) -> void* {
        void* p = ws + off;
        off = (off + bytes + 255) & ~(size_t)255;
        return p;
    };
    float*    h    = (float*)   alloc((size_t)SEQ * DMODEL * 4);
    _Float16* xn   = (_Float16*)alloc((size_t)SEQ * DMODEL * 2);
    float*    qkv  = (float*)   alloc((size_t)SEQ * QKVO * 4);
    _Float16* Qf   = (_Float16*)alloc((size_t)NHEAD * SEQ * HDIM * 2);
    _Float16* Kf   = (_Float16*)alloc((size_t)NKVH * SEQ * HDIM * 2);
    _Float16* Vf   = (_Float16*)alloc((size_t)NKVH * SEQ * HDIM * 2);
    _Float16* ctx  = (_Float16*)alloc((size_t)SEQ * DMODEL * 2);
    _Float16* gt   = (_Float16*)alloc((size_t)SEQ * FFH * 2);
    // scores (33.5MB) and ff (56MB) are not simultaneously live -> share region
    void* bigRegion = alloc((size_t)SEQ * FF2H * 4);
    float* sc = (float*)bigRegion;
    float* ff = (float*)bigRegion;

    embed_gather<<<SEQ, 256, 0, stream>>>(ids, emb, h);

    for (int l = 0; l < 2; ++l) {
        rmsnorm_f16<<<SEQ, 256, 0, stream>>>(h, ln1 + (size_t)l * DMODEL, xn);

        gemm_q8_wmma<true, false><<<dim3(QKVO / 64, SEQ / 64), 256, 0, stream>>>(
            xn, qkv_q + (size_t)l * DMODEL * QKVO,
            qkv_s + (size_t)l * (DMODEL / GSZ) * QKVO,
            qkv_z + (size_t)l * (DMODEL / GSZ) * QKVO,
            qkv_b + (size_t)l * QKVO, nullptr, qkv, SEQ, QKVO, DMODEL);

        rope_split<<<dim3(36, SEQ), 64, 0, stream>>>(qkv, Qf, Kf, Vf);

        attn_scores_wmma<<<dim3(SEQ / 64, SEQ / 64, NHEAD), 256, 0, stream>>>(Qf, Kf, sc);
        softmax512<<<NHEAD * SEQ, 256, 0, stream>>>(sc);
        attn_pv_wmma<<<dim3(1, SEQ / 64, NHEAD), 256, 0, stream>>>(sc, Vf, ctx);

        gemm_q8_wmma<false, true><<<dim3(DMODEL / 64, SEQ / 64), 256, 0, stream>>>(
            ctx, dn_q + (size_t)l * DMODEL * DMODEL,
            dn_s + (size_t)l * (DMODEL / GSZ) * DMODEL,
            dn_z + (size_t)l * (DMODEL / GSZ) * DMODEL,
            nullptr, h, h, SEQ, DMODEL, DMODEL);

        rmsnorm_f16<<<SEQ, 256, 0, stream>>>(h, ln2 + (size_t)l * DMODEL, xn);

        gemm_q8_wmma<false, false><<<dim3(FF2H / 64, SEQ / 64), 256, 0, stream>>>(
            xn, h4_q + (size_t)l * DMODEL * FF2H,
            h4_s + (size_t)l * (DMODEL / GSZ) * FF2H,
            h4_z + (size_t)l * (DMODEL / GSZ) * FF2H,
            nullptr, nullptr, ff, SEQ, FF2H, DMODEL);

        silu_gate<<<(int)(((size_t)SEQ * FFH + 255) / 256), 256, 0, stream>>>(ff, gt);

        gemm_q8_wmma<false, true><<<dim3(DMODEL / 64, SEQ / 64), 256, 0, stream>>>(
            gt, fh_q + (size_t)l * FFH * DMODEL,
            fh_s + (size_t)l * (FFH / GSZ) * DMODEL,
            fh_z + (size_t)l * (FFH / GSZ) * DMODEL,
            nullptr, h, h, SEQ, DMODEL, FFH);
    }

    rmsnorm_f16<<<SEQ, 256, 0, stream>>>(h, lnf, xn);

    gemm_f32w_wmma<<<dim3(VOCAB / 64, SEQ / 64), 256, 0, stream>>>(
        xn, outw, (float*)d_out, SEQ, VOCAB, DMODEL);
}
